// tramsform_conv_58342835749296
// MI455X (gfx1250) — matrix operand order repaired and checked by reading, not compiled, not run on hardware
//
#include <hip/hip_runtime.h>
#include <stdint.h>

// CDNA5 / gfx1250: wave32, WMMA 16x16x32 bf16 -> f32 accum.
typedef __attribute__((ext_vector_type(16))) __bf16 v16bf;
typedef __attribute__((ext_vector_type(8)))  float  v8f;
typedef __attribute__((ext_vector_type(4)))  float  f4;
typedef __attribute__((ext_vector_type(4)))  __bf16 bf4;

#define HH       128
#define WW       128
#define L_IMG    (HH * WW)
#define C_IN     64
#define O_OUT    64
#define CKK      576          // 9 * 64, contraction dim, ordered ck' = k*64 + c
#define TILE_L   64           // output columns per workgroup
#define LSTRIDE  580          // bf16/row = 290 dwords = 145 qwords; qword column
                              // access conflict-free: banks 2*(17*l mod 32)+{0,1}
#define NTHREADS 128
#define KSTEPS   18           // 576 / 32
#define WS_BYTES (O_OUT * LSTRIDE * 2)   // 74240 B: pre-converted W in d_ws

__device__ __forceinline__ uint16_t f32_to_bf16(float f) {
    return __builtin_bit_cast(uint16_t, (__bf16)f);   // native v_cvt_bf16_f32 (RNE)
}

__device__ __forceinline__ uint64_t pack4_bf16(f4 v) {
    return __builtin_bit_cast(uint64_t, __builtin_convertvector(v, bf4));
}

union Frag {
    uint64_t q[4];
    uint32_t u[8];
    v16bf    v;
};

// ---- pre-kernel: W [C,3,3,O] f32 -> d_ws bf16 [o][ck'=k*64+c], padded stride ----
__global__ __launch_bounds__(128)
void convert_W(const float* __restrict__ Wfl, uint16_t* __restrict__ Wbf) {
    const int o = blockIdx.x;                       // 0..63
    for (int ckp = threadIdx.x; ckp < LSTRIDE; ckp += 128) {
        uint16_t v = 0;
        if (ckp < CKK) {
            const int k = ckp >> 6, c = ckp & 63;
            v = f32_to_bf16(Wfl[(c * 9 + k) * O_OUT + o]);
        }
        Wbf[o * LSTRIDE + ckp] = v;                 // contiguous b16 stores
    }
}

__global__ __launch_bounds__(NTHREADS)
void transform_conv_wmma(const float* __restrict__ feat,   // [B, C, H, W]
                         const float* __restrict__ tmat,   // [B, 9, H*W]
                         const float* __restrict__ Wfl,    // [(c*9+k)*64+o] f32
                         const uint16_t* __restrict__ Wbf, // pre-converted, or null
                         float* __restrict__ out) {        // [B, O, H*W]
    __shared__ __align__(16) uint16_t Wlds[O_OUT * LSTRIDE];   // Wlds[o][ck']
    __shared__ __align__(16) uint16_t Qlds[TILE_L * LSTRIDE];  // Qlds[l][ck']
    __shared__ float Tlds[9 * TILE_L];                         // trans[k][l-tile]

    const int tid = threadIdx.x;
    const int wg  = blockIdx.x;
    const int j0  = (wg & 1) * TILE_L;   // half-row start
    const int i   = (wg >> 1) & 127;     // image row
    const int b   = wg >> 8;             // batch

    // ---- stage trans: Tlds[k][l] = tmat[b, k, i*W + j0 + l] ----
    for (int s = tid; s < 9 * TILE_L; s += NTHREADS) {
        const int k = s >> 6, l = s & 63;
        Tlds[k * TILE_L + l] = tmat[(b * 9 + k) * L_IMG + i * WW + j0 + l];
    }

    // ---- stage W: fast b128 copy from pre-converted d_ws, else convert here ----
    if (Wbf) {
        const uint4* __restrict__ src = (const uint4*)Wbf;
        uint4* dst = (uint4*)Wlds;
        for (int s = tid; s < WS_BYTES / 16; s += NTHREADS)    // 4640 chunks
            dst[s] = src[s];
    } else {
        // fallback: thread owns row o, half of ck'; 4-wide packed b64 stores
        const int o    = tid & 63;
        const int half = tid >> 6;
        uint64_t* wrow = (uint64_t*)&Wlds[o * LSTRIDE + half * 288];
        for (int cq = 0; cq < 72; ++cq) {
            f4 v;
#pragma unroll
            for (int e = 0; e < 4; ++e) {
                const int ckp = half * 288 + cq * 4 + e;
                const int k = ckp >> 6, c = ckp & 63;
                v[e] = Wfl[(c * 9 + k) * O_OUT + o];
            }
            wrow[cq] = pack4_bf16(v);
        }
    }
    __syncthreads();

    // ---- build Q tile: Qlds[l][k*64+c] = feat[b,c,i+ki-1, j0+l+kj-1]*trans[k][l] ----
    // Per-k constants hoisted; inner: 4 coalesced loads -> packed cvt -> ds_store_b64.
    {
        const int sub = tid >> 6;        // channel half
        const int l   = tid & 63;
        for (int k = 0; k < 9; ++k) {
            const int ki  = (k * 11) >> 5;       // k/3 for k in 0..8
            const int kj  = k - ki * 3;
            const int row = i + ki - 1;
            const int col = j0 + l + kj - 1;
            const bool ok = ((unsigned)row < (unsigned)HH) &&
                            ((unsigned)col < (unsigned)WW);
            const float tval = Tlds[k * TILE_L + l];
            const float* __restrict__ fp =
                feat + (((b * C_IN + sub * 32) * HH + row) * WW + col);
            uint64_t* __restrict__ qp64 =
                (uint64_t*)&Qlds[l * LSTRIDE + k * 64 + sub * 32];
#pragma unroll
            for (int cq = 0; cq < 8; ++cq) {
                f4 v;
#pragma unroll
                for (int e = 0; e < 4; ++e)
                    v[e] = (ok ? fp[(cq * 4 + e) * L_IMG] : 0.0f) * tval;
                qp64[cq] = pack4_bf16(v);
            }
        }
    }
    __syncthreads();

    // ---- WMMA main loop: out(64x64) = Wlds (64x576) x Qlds^T (576x64) ----
    const int lane = tid & 31;
    const int wv   = tid >> 5;            // wave 0..3
    const int wm   = (wv >> 1) * 32;      // output-row (o) base
    const int wn   = (wv & 1) * 32;       // output-col (l) base
    const int tn   = lane & 15;
    const int th   = lane >> 4;           // lane half -> K sub-offset

    const uint64_t* __restrict__ W64 = (const uint64_t*)Wlds;
    const uint64_t* __restrict__ Q64 = (const uint64_t*)Qlds;

    v8f acc[2][2] = {};

    for (int ks = 0; ks < KSTEPS; ++ks) {
        Frag afr[2], bfr[2];
        // A (16x32 bf16): VGPR p -> K = (p>>2)*16 + (p&3)*2 + th*8; qword pp = p>>1
#pragma unroll
        for (int mf = 0; mf < 2; ++mf) {
            const int mrow = (wm + mf * 16 + tn) * 145;
#pragma unroll
            for (int pp = 0; pp < 4; ++pp)
                afr[mf].q[pp] =
                    W64[mrow + ks * 8 + (pp >> 1) * 4 + th * 2 + (pp & 1)];
        }
        // B (32x16 bf16): same K layout as A: qword pp -> K = (pp>>1)*16 + th*8 + (pp&1)*4
#pragma unroll
        for (int nf = 0; nf < 2; ++nf) {
            const int lrow = (wn + nf * 16 + tn) * 145;
#pragma unroll
            for (int pp = 0; pp < 4; ++pp)
                bfr[nf].q[pp] = Q64[lrow + ks * 8 + (pp >> 1) * 4 + th * 2 + (pp & 1)];
        }
#pragma unroll
        for (int mf = 0; mf < 2; ++mf)
#pragma unroll
            for (int nf = 0; nf < 2; ++nf)
                acc[mf][nf] = __builtin_amdgcn_wmma_f32_16x16x32_bf16(
                    /*neg_a=*/false, afr[mf].v,
                    /*neg_b=*/false, bfr[nf].v,
                    /*c_mod=*/(short)0, acc[mf][nf],
                    /*reuse_a=*/false, /*reuse_b=*/false);
    }

    // ---- store f32 C/D layout: VGPR v -> M = v + th*8; lane tn -> N ----
    const int lg0 = i * WW + j0;
#pragma unroll
    for (int mf = 0; mf < 2; ++mf) {
#pragma unroll
        for (int nf = 0; nf < 2; ++nf) {
            const int l = wn + nf * 16 + tn;
#pragma unroll
            for (int v = 0; v < 8; ++v) {
                const int m = wm + mf * 16 + (th << 3) + v;
                out[(b * O_OUT + m) * L_IMG + lg0 + l] = acc[mf][nf][v];
            }
        }
    }
}

extern "C" void kernel_launch(void* const* d_in, const int* in_sizes, int n_in,
                              void* d_out, int out_size, void* d_ws, size_t ws_size,
                              hipStream_t stream) {
    const float* feat = (const float*)d_in[0];  // [8, 64, 128, 128] f32
    const float* tmat = (const float*)d_in[1];  // [8, 9, 16384]     f32
    const float* Wfl  = (const float*)d_in[2];  // [64, 3, 3, 64]    f32
    float* out = (float*)d_out;                 // [8, 64, 128, 128] f32

    const bool use_ws = (d_ws != nullptr) && (ws_size >= (size_t)WS_BYTES);
    uint16_t* Wbf = use_ws ? (uint16_t*)d_ws : nullptr;

    if (use_ws)
        hipLaunchKernelGGL(convert_W, dim3(O_OUT), dim3(128), 0, stream, Wfl, Wbf);

    // grid: B * H * (W / TILE_L) = 8 * 128 * 2 = 2048 workgroups, 4 waves each
    hipLaunchKernelGGL(transform_conv_wmma, dim3(2048), dim3(NTHREADS), 0, stream,
                       feat, tmat, Wfl, Wbf, out);
}
